// OnegateMixtureOfExperts_49134425866510
// MI455X (gfx1250) — matrix operand order repaired and checked by reading, not compiled, not run on hardware
//
#include <hip/hip_runtime.h>
#include <hip/hip_bf16.h>

// Problem constants (from reference)
#define B_SZ   16384
#define D_IN   1024
#define D_EXPD 1024
#define D_TOW  512
#define N_EXP  8
#define N_TASK 4

// LDS tile row stride in halves (32 K + 8 pad to spread banks; keeps 16B align)
#define LDW 40

typedef __attribute__((ext_vector_type(16))) __bf16       bf16x16;
typedef __attribute__((ext_vector_type(4)))  __bf16       bf16x4;
typedef __attribute__((ext_vector_type(8)))  float        f32x8;
typedef __attribute__((ext_vector_type(4)))  float        f32x4;
typedef __attribute__((ext_vector_type(4)))  unsigned int u32x4;

union FragAB { bf16x16 v; u32x4 q[2]; };

// ---------------------------------------------------------------------------
// gfx1250 async global->LDS helpers (inline asm; ASYNCcnt-tracked)
// ---------------------------------------------------------------------------
__device__ __forceinline__ unsigned lds_addr_of(const void* p) {
    // generic -> LDS(addrspace 3) pointer, then to byte offset within wave LDS
    return (unsigned)(unsigned long long)(const __attribute__((address_space(3))) void*)p;
}

__device__ __forceinline__ void cp_async_b128(unsigned lds_byte, const void* gptr) {
    asm volatile("global_load_async_to_lds_b128 %0, %1, off"
                 :: "v"(lds_byte), "v"(gptr) : "memory");
}

__device__ __forceinline__ void wait_async0() {
    asm volatile("s_wait_asynccnt 0x0" ::: "memory");
}

// Copy a 128-row x 32-half bf16 tile (64B/row) global->LDS, 4 x b128 per row.
// 256 threads issue 2 chunks each (512 chunks total).
__device__ __forceinline__ void stage128x32(unsigned ldsBase, const __bf16* g, int ldh) {
#pragma unroll
    for (int j = 0; j < 2; ++j) {
        const int c   = (int)threadIdx.x + j * 256;
        const int row = c >> 2;
        const int q   = c & 3;
        cp_async_b128(ldsBase + (unsigned)(row * (LDW * 2) + q * 16),
                      g + (size_t)row * ldh + q * 8);
    }
}

// ---------------------------------------------------------------------------
// Pre-pass: f32 -> bf16 straight convert (for x)
// ---------------------------------------------------------------------------
__global__ __launch_bounds__(256) void cvt_kernel(const float* __restrict__ in,
                                                  __bf16* __restrict__ out) {
    const size_t i = ((size_t)blockIdx.x * 256 + threadIdx.x) * 4;
    f32x4 f = *(const f32x4*)(in + i);
    bf16x4 h;
    h[0] = (__bf16)f[0]; h[1] = (__bf16)f[1];
    h[2] = (__bf16)f[2]; h[3] = (__bf16)f[3];
    *(bf16x4*)(out + i) = h;
}

// ---------------------------------------------------------------------------
// Pre-pass: batched transpose + convert: in [bz][R][C] f32 -> out [bz][C][R] bf16
// ---------------------------------------------------------------------------
__global__ __launch_bounds__(256) void transpose_cvt_kernel(const float* __restrict__ in,
                                                            __bf16* __restrict__ out,
                                                            int R, int C) {
    __shared__ __bf16 tile[32][33];
    const float* src = in  + (size_t)blockIdx.z * R * C;
    __bf16*      dst = out + (size_t)blockIdx.z * R * C;
    const int tx = threadIdx.x & 31;
    const int ty = threadIdx.x >> 5;            // 32 x 8
#pragma unroll
    for (int j = 0; j < 32; j += 8) {
        const int r = blockIdx.y * 32 + ty + j;
        const int c = blockIdx.x * 32 + tx;
        tile[ty + j][tx] = (__bf16)src[(size_t)r * C + c];
    }
    __syncthreads();
#pragma unroll
    for (int j = 0; j < 32; j += 8) {
        const int c = blockIdx.x * 32 + ty + j;
        const int r = blockIdx.y * 32 + tx;
        dst[(size_t)c * R + r] = tile[tx][ty + j];
    }
}

// ---------------------------------------------------------------------------
// Kernel 0: gates = x @ gate_w   [B, 8]  (0.27 GFLOP — scalar is fine)
// ---------------------------------------------------------------------------
__global__ __launch_bounds__(256) void gates_kernel(const float* __restrict__ x,
                                                    const float* __restrict__ gw,
                                                    float* __restrict__ gates) {
    const int idx = blockIdx.x * 256 + threadIdx.x;
    const int b = idx >> 3;
    const int e = idx & 7;
    const float* xr = x + (size_t)b * D_IN;
    float acc = 0.f;
#pragma unroll 8
    for (int i = 0; i < D_IN; ++i)
        acc = fmaf(xr[i], gw[i * N_EXP + e], acc);
    gates[idx] = acc;
}

// ---------------------------------------------------------------------------
// Kernel 1: shared[b,d] = sum_e gates[b,e] * relu(x @ exp_w[e] + exp_b[e])
// 128x128 tile / WG, 8 waves (wave tile 32x64 = 2x4 WMMA subtiles).
// Double-buffered async staging pipelined across the flattened (e,kc) loop.
// ---------------------------------------------------------------------------
__global__ __launch_bounds__(256) void moe_expert_kernel(
    const __bf16* __restrict__ xb, const __bf16* __restrict__ ewT,
    const float* __restrict__ exp_b, const float* __restrict__ gates,
    __bf16* __restrict__ shb) {

    __shared__ __align__(16) __bf16 lA[2][128 * LDW];   // x tile  [m][k]
    __shared__ __align__(16) __bf16 lB[2][128 * LDW];   // wT tile [n][k]

    const int tid   = threadIdx.x;
    const int lane  = tid & 31;
    const int wave  = tid >> 5;
    const int waveM = wave & 3;
    const int waveN = wave >> 2;
    const int half  = lane >> 4;
    const int l16   = lane & 15;

    const int rowBase = blockIdx.x * 128;
    const int colBase = blockIdx.y * 128;

    const unsigned offA[2] = { lds_addr_of(&lA[0][0]), lds_addr_of(&lA[1][0]) };
    const unsigned offB[2] = { lds_addr_of(&lB[0][0]), lds_addr_of(&lB[1][0]) };

    const __bf16* xbase = xb + (size_t)rowBase * D_IN;

    f32x8 sacc[2][4], acc[2][4];
#pragma unroll
    for (int mt = 0; mt < 2; ++mt)
#pragma unroll
        for (int nt = 0; nt < 4; ++nt)
#pragma unroll
            for (int i = 0; i < 8; ++i) { sacc[mt][nt][i] = 0.f; acc[mt][nt][i] = 0.f; }

    const int NSTEP = N_EXP * (D_IN / 32);

    // issue async copies for step s into buffer s&1
    auto issue = [&](int s) {
        const int e   = s >> 5;
        const int k0  = (s & 31) * 32;
        const int buf = s & 1;
        stage128x32(offA[buf], xbase + k0, D_IN);
        stage128x32(offB[buf], ewT + ((size_t)e * D_EXPD + colBase) * D_IN + k0, D_IN);
    };

    issue(0);
    for (int step = 0; step < NSTEP; ++step) {
        wait_async0();
        __syncthreads();                 // current buffer ready, prior compute done
        if (step + 1 < NSTEP) issue(step + 1);   // overlap next copies with WMMA
        const int buf = step & 1;

        FragAB fa[2];
#pragma unroll
        for (int mt = 0; mt < 2; ++mt) {
            const int row = waveM * 32 + mt * 16 + l16;
            // halves 0..7 : K = 8*half + j ; halves 8..15 : K = 16 + 8*half + j
            fa[mt].q[0] = *(const u32x4*)(&lA[buf][row * LDW + 8 * half]);
            fa[mt].q[1] = *(const u32x4*)(&lA[buf][row * LDW + 16 + 8 * half]);
        }
#pragma unroll
        for (int nt = 0; nt < 4; ++nt) {
            const int col = waveN * 64 + nt * 16 + l16;
            FragAB fb;   // halves 0..7 : K = 8*half + j ; halves 8..15 : K = 16 + 8*half + j
            fb.q[0] = *(const u32x4*)(&lB[buf][col * LDW + 8 * half]);
            fb.q[1] = *(const u32x4*)(&lB[buf][col * LDW + 16 + 8 * half]);
            acc[0][nt] = __builtin_amdgcn_wmma_f32_16x16x32_bf16(
                false, fa[0].v, false, fb.v, (short)0, acc[0][nt], false, false);
            acc[1][nt] = __builtin_amdgcn_wmma_f32_16x16x32_bf16(
                false, fa[1].v, false, fb.v, (short)0, acc[1][nt], false, false);
        }

        if ((step & 31) == 31) {         // expert complete: fold gate*relu(acc+bias)
            const int e = step >> 5;
            float g[2][8];
#pragma unroll
            for (int mt = 0; mt < 2; ++mt)
#pragma unroll
                for (int i = 0; i < 8; ++i)
                    g[mt][i] = gates[(size_t)(rowBase + waveM * 32 + mt * 16 + half * 8 + i) * N_EXP + e];
            float bv[4];
#pragma unroll
            for (int nt = 0; nt < 4; ++nt)
                bv[nt] = exp_b[(size_t)e * D_EXPD + colBase + waveN * 64 + nt * 16 + l16];
#pragma unroll
            for (int mt = 0; mt < 2; ++mt)
#pragma unroll
                for (int nt = 0; nt < 4; ++nt)
#pragma unroll
                    for (int i = 0; i < 8; ++i) {
                        float v = fmaxf(acc[mt][nt][i] + bv[nt], 0.f);
                        sacc[mt][nt][i] = fmaf(g[mt][i], v, sacc[mt][nt][i]);
                        acc[mt][nt][i] = 0.f;
                    }
        }
    }

    // Store shared tile as bf16 (tower consumes bf16 anyway)
#pragma unroll
    for (int mt = 0; mt < 2; ++mt)
#pragma unroll
        for (int nt = 0; nt < 4; ++nt)
#pragma unroll
            for (int i = 0; i < 8; ++i) {
                const int row = rowBase + waveM * 32 + mt * 16 + half * 8 + i;
                const int col = colBase + waveN * 64 + nt * 16 + l16;
                shb[(size_t)row * D_EXPD + col] = (__bf16)sacc[mt][nt][i];
            }
}

// ---------------------------------------------------------------------------
// Kernel 2: out[t,b] = tb2[t] + sum_h tw2[t,h] * relu(shared[b] @ tw1[t,:,h] + tb1[t,h])
// Same pipeline; fuses GEMM + relu + final dot, reducing N via shuffles + LDS.
// ---------------------------------------------------------------------------
__global__ __launch_bounds__(256) void tower_kernel(
    const __bf16* __restrict__ shb, const __bf16* __restrict__ t1T,
    const float* __restrict__ tb1,  const float* __restrict__ tw2,
    const float* __restrict__ tb2,  float* __restrict__ out) {

    __shared__ __align__(16) __bf16 lA[2][128 * LDW];
    __shared__ __align__(16) __bf16 lB[2][128 * LDW];
    __shared__ float red[128];

    const int tid   = threadIdx.x;
    const int lane  = tid & 31;
    const int wave  = tid >> 5;
    const int waveM = wave & 3;
    const int waveN = wave >> 2;
    const int half  = lane >> 4;
    const int l16   = lane & 15;

    const int rowBase = blockIdx.x * 128;
    const int t       = blockIdx.y;

    const unsigned offA[2] = { lds_addr_of(&lA[0][0]), lds_addr_of(&lA[1][0]) };
    const unsigned offB[2] = { lds_addr_of(&lB[0][0]), lds_addr_of(&lB[1][0]) };

    const __bf16* abase = shb + (size_t)rowBase * D_EXPD;
    const __bf16* bbase = t1T + (size_t)t * D_TOW * D_EXPD;   // [h][d]

    float rowpart[2][8];
#pragma unroll
    for (int mt = 0; mt < 2; ++mt)
#pragma unroll
        for (int i = 0; i < 8; ++i) rowpart[mt][i] = 0.f;

    f32x8 acc[2][4];
#pragma unroll
    for (int mt = 0; mt < 2; ++mt)
#pragma unroll
        for (int nt = 0; nt < 4; ++nt)
#pragma unroll
            for (int i = 0; i < 8; ++i) acc[mt][nt][i] = 0.f;

    const int NSTEP = (D_TOW / 128) * (D_EXPD / 32);   // 4 * 32

    auto issue = [&](int s) {
        const int hb  = s >> 5;
        const int k0  = (s & 31) * 32;
        const int buf = s & 1;
        stage128x32(offA[buf], abase + k0, D_EXPD);
        stage128x32(offB[buf], bbase + (size_t)(hb * 128) * D_EXPD + k0, D_EXPD);
    };

    issue(0);
    for (int step = 0; step < NSTEP; ++step) {
        wait_async0();
        __syncthreads();
        if (step + 1 < NSTEP) issue(step + 1);
        const int buf = step & 1;

        FragAB fa[2];
#pragma unroll
        for (int mt = 0; mt < 2; ++mt) {
            const int row = waveM * 32 + mt * 16 + l16;
            fa[mt].q[0] = *(const u32x4*)(&lA[buf][row * LDW + 8 * half]);
            fa[mt].q[1] = *(const u32x4*)(&lA[buf][row * LDW + 16 + 8 * half]);
        }
#pragma unroll
        for (int nt = 0; nt < 4; ++nt) {
            const int col = waveN * 64 + nt * 16 + l16;
            FragAB fb;
            fb.q[0] = *(const u32x4*)(&lB[buf][col * LDW + 8 * half]);
            fb.q[1] = *(const u32x4*)(&lB[buf][col * LDW + 16 + 8 * half]);
            acc[0][nt] = __builtin_amdgcn_wmma_f32_16x16x32_bf16(
                false, fa[0].v, false, fb.v, (short)0, acc[0][nt], false, false);
            acc[1][nt] = __builtin_amdgcn_wmma_f32_16x16x32_bf16(
                false, fa[1].v, false, fb.v, (short)0, acc[1][nt], false, false);
        }

        if ((step & 31) == 31) {   // h-block complete: rowpart += tw2[h]*relu(acc+tb1[h])
            const int hBase = (step >> 5) * 128;
#pragma unroll
            for (int nt = 0; nt < 4; ++nt) {
                const int h = hBase + waveN * 64 + nt * 16 + l16;
                const float b1 = tb1[t * D_TOW + h];
                const float w2 = tw2[t * D_TOW + h];
#pragma unroll
                for (int mt = 0; mt < 2; ++mt)
#pragma unroll
                    for (int i = 0; i < 8; ++i) {
                        rowpart[mt][i] = fmaf(w2, fmaxf(acc[mt][nt][i] + b1, 0.f), rowpart[mt][i]);
                        acc[mt][nt][i] = 0.f;
                    }
            }
        }
    }

    // Reduce across the 16 N-lanes of each half, then across the 2 N-waves via LDS
    if (tid < 128) red[tid] = 0.f;
    __syncthreads();
#pragma unroll
    for (int mt = 0; mt < 2; ++mt)
#pragma unroll
        for (int i = 0; i < 8; ++i) {
            float v = rowpart[mt][i];
            v += __shfl_xor(v, 1, 32);
            v += __shfl_xor(v, 2, 32);
            v += __shfl_xor(v, 4, 32);
            v += __shfl_xor(v, 8, 32);
            if (l16 == 0)
                atomicAdd(&red[waveM * 32 + mt * 16 + half * 8 + i], v);
        }
    __syncthreads();
    if (tid < 128)
        out[(size_t)t * B_SZ + rowBase + tid] = red[tid] + tb2[t];
}

// ---------------------------------------------------------------------------
extern "C" void kernel_launch(void* const* d_in, const int* in_sizes, int n_in,
                              void* d_out, int out_size, void* d_ws, size_t ws_size,
                              hipStream_t stream) {
    const float* x    = (const float*)d_in[0];
    const float* gw   = (const float*)d_in[1];
    const float* expw = (const float*)d_in[2];
    const float* expb = (const float*)d_in[3];
    const float* tw1  = (const float*)d_in[4];
    const float* tb1  = (const float*)d_in[5];
    const float* tw2  = (const float*)d_in[6];
    const float* tb2  = (const float*)d_in[7];
    float* out = (float*)d_out;

    // Workspace carve-up
    float*  gates = (float*)d_ws;                                   // [B,8]          0.5 MB
    __bf16* xb    = (__bf16*)(gates + (size_t)B_SZ * N_EXP);        // [B,D_IN]       32 MB
    __bf16* ewT   = xb  + (size_t)B_SZ * D_IN;                      // [E,D_EXP,D_IN] 16 MB
    __bf16* t1T   = ewT + (size_t)N_EXP * D_EXPD * D_IN;            // [T,D_TOW,D_EXP] 4 MB
    __bf16* shb   = t1T + (size_t)N_TASK * D_TOW * D_EXPD;          // [B,D_EXP]      32 MB

    // Pre-pass: convert / transpose weights and activations to bf16
    cvt_kernel<<<(B_SZ * D_IN) / (256 * 4), 256, 0, stream>>>(x, xb);
    transpose_cvt_kernel<<<dim3(D_EXPD / 32, D_IN / 32, N_EXP), 256, 0, stream>>>(
        expw, ewT, D_IN, D_EXPD);
    transpose_cvt_kernel<<<dim3(D_TOW / 32, D_EXPD / 32, N_TASK), 256, 0, stream>>>(
        tw1, t1T, D_EXPD, D_TOW);
    gates_kernel<<<(B_SZ * N_EXP) / 256, 256, 0, stream>>>(x, gw, gates);

    dim3 g1(B_SZ / 128, D_EXPD / 128);
    moe_expert_kernel<<<g1, 256, 0, stream>>>(xb, ewT, expb, gates, shb);

    dim3 g2(B_SZ / 128, N_TASK);
    tower_kernel<<<g2, 256, 0, stream>>>(shb, t1T, tb1, tw2, tb2, out);
}
